// HeteroGNN_1322849927480
// MI455X (gfx1250) — compile-verified, matrix operand order checked by reading, not run on hardware
//
#include <hip/hip_runtime.h>
#include <hip/hip_bf16.h>
#include <stdint.h>
#include <stddef.h>

typedef __bf16 bf16_t;
typedef __attribute__((ext_vector_type(16))) __bf16 v16bf;
typedef __attribute__((ext_vector_type(8)))  __bf16 v8bf;
typedef __attribute__((ext_vector_type(8)))  float  v8f;

union V16U { v16bf v; v8bf h[2]; };

// ---------------- elementwise / prep kernels ----------------

__global__ __launch_bounds__(256) void k_zero_f32(float* __restrict__ p, int n) {
  int i = blockIdx.x * 256 + threadIdx.x;
  if (i < n) p[i] = 0.0f;
}

__global__ __launch_bounds__(256) void k_cvt_bf16(const float* __restrict__ s,
                                                  bf16_t* __restrict__ d, int n) {
  int i = blockIdx.x * 256 + threadIdx.x;
  if (i < n) d[i] = (bf16_t)s[i];
}

// W: [nmat][128(k)][128(n)] f32  ->  WT: [nmat][128(n)][128(k)] bf16
__global__ __launch_bounds__(256) void k_pack_wt(const float* __restrict__ W,
                                                 bf16_t* __restrict__ WT, int total) {
  int i = blockIdx.x * 256 + threadIdx.x;
  if (i >= total) return;
  int m = i >> 14, rem = i & 16383;
  int n = rem >> 7, k = rem & 127;
  WT[i] = (bf16_t)W[(m << 14) + (k << 7) + n];
}

__global__ __launch_bounds__(256) void k_count(const int* __restrict__ idx,
                                               float* __restrict__ deg, int E) {
  int i = blockIdx.x * 256 + threadIdx.x;
  if (i < E) atomicAdd(&deg[idx[i]], 1.0f);
}

__global__ __launch_bounds__(256) void k_dinv(float* __restrict__ deg, int n, float selfadd) {
  int i = blockIdx.x * 256 + threadIdx.x;
  if (i >= n) return;
  float d = deg[i] + selfadd;
  deg[i] = (d > 0.0f) ? rsqrtf(fmaxf(d, 1.0f)) : 0.0f;
}

// ---------------- WMMA GEMM: C[MxN] = A[Mx128] * BT[Nx128]^T (+bias) ----------------
// A row-major bf16 (lda=128), BT holds B transposed (each row = one output column's
// 128 K-weights), C row-major f32 with stride ldc.
__global__ __launch_bounds__(256) void k_gemm_bf16(
    const bf16_t* __restrict__ A, const bf16_t* __restrict__ BT,
    float* __restrict__ C, const float* __restrict__ bias,
    int tilesM, int tilesN, int ldc) {
  int wv   = (int)((blockIdx.x * 256u + threadIdx.x) >> 5);
  int lane = threadIdx.x & 31;
  if (wv >= tilesM * tilesN) return;
  int tm = wv / tilesN, tn = wv - tm * tilesN;
  int sub = lane & 15;
  int k0  = (lane & 16) ? 8 : 0;   // A/B 16-bit operand layout: lanes 16-31 hold K+8/K+24
  const bf16_t* pa = A  + (size_t)((tm << 4) + sub) * 128 + k0;
  const bf16_t* pb = BT + (size_t)((tn << 4) + sub) * 128 + k0;
  v8f acc = {};
#pragma unroll
  for (int kc = 0; kc < 128; kc += 32) {
    V16U a, b;
    a.h[0] = *(const v8bf*)(pa + kc);
    a.h[1] = *(const v8bf*)(pa + kc + 16);
    b.h[0] = *(const v8bf*)(pb + kc);
    b.h[1] = *(const v8bf*)(pb + kc + 16);
    acc = __builtin_amdgcn_wmma_f32_16x16x32_bf16(
        /*neg_a=*/false, a.v, /*neg_b=*/false, b.v,
        /*c_mod=*/(short)0, acc, /*reuse_a=*/false, /*reuse_b=*/false);
  }
  int n  = (tn << 4) + sub;
  int m0 = (tm << 4) + ((lane >> 4) << 3);
  float badd = bias ? bias[n] : 0.0f;
#pragma unroll
  for (int i = 0; i < 8; ++i)
    C[(size_t)(m0 + i) * ldc + n] = acc[i] + badd;
}

// ---------------- edge scatter: one wave32 per edge ----------------
__global__ __launch_bounds__(256) void k_scatter(
    const int* __restrict__ row, const int* __restrict__ col,
    const float* __restrict__ h, float* __restrict__ acc,
    const float* __restrict__ dinv_s, const float* __restrict__ dinv_d, int E) {
  int wv   = (int)((blockIdx.x * 256u + threadIdx.x) >> 5);
  int lane = threadIdx.x & 31;
  if (wv >= E) return;
  int r = row[wv], c = col[wv];
  float norm = dinv_s[r] * dinv_d[c];
  const float4* hv = (const float4*)(h + (size_t)r * 128);
  float4 v = hv[lane];
  float* ad = acc + (size_t)c * 128 + lane * 4;
  atomicAdd(ad + 0, v.x * norm);
  atomicAdd(ad + 1, v.y * norm);
  atomicAdd(ad + 2, v.z * norm);
  atomicAdd(ad + 3, v.w * norm);
}

__global__ __launch_bounds__(256) void k_selfloop(
    const float* __restrict__ h, float* __restrict__ acc,
    const float* __restrict__ dinv_s, const float* __restrict__ dinv_d, int total) {
  int i = blockIdx.x * 256 + threadIdx.x;
  if (i >= total) return;
  int node = i >> 7;
  acc[i] += h[i] * dinv_s[node] * dinv_d[node];
}

__global__ __launch_bounds__(256) void k_relu_bias(
    float* __restrict__ acc, const float* __restrict__ b1,
    const float* __restrict__ b2, int total) {
  int i = blockIdx.x * 256 + threadIdx.x;
  if (i >= total) return;
  int k = i & 127;
  acc[i] = fmaxf(acc[i] + b1[k] + b2[k], 0.0f);
}

// ---------------- host orchestration ----------------

extern "C" void kernel_launch(void* const* d_in, const int* in_sizes, int n_in,
                              void* d_out, int out_size, void* d_ws, size_t ws_size,
                              hipStream_t stream) {
  (void)n_in; (void)out_size; (void)ws_size;
  const float* x0[2]  = {(const float*)d_in[0], (const float*)d_in[1]};
  const int*   ei[4]  = {(const int*)d_in[2], (const int*)d_in[3],
                         (const int*)d_in[4], (const int*)d_in[5]};
  const float* Ws     = (const float*)d_in[6];
  const float* bs     = (const float*)d_in[7];
  const float* lin_w  = (const float*)d_in[8];
  const float* lin_b  = (const float*)d_in[9];
  float* out = (float*)d_out;

  const int H = 128;
  const int Nn[2] = { in_sizes[0] / H, in_sizes[1] / H };   // drug, dis node counts
  const int E     = in_sizes[2] / 2;
  const int OUT   = in_sizes[8] / H;
  const int Lnum  = in_sizes[6] / (4 * H * H);

  // workspace carve-out (256B aligned)
  size_t off = 0; char* base = (char*)d_ws;
  auto alloc = [&](size_t bytes) -> void* {
    void* p = base + off;
    off = (off + bytes + 255) & ~(size_t)255;
    return p;
  };
  bf16_t* xb[2];  float* acc[2];
  xb[0]  = (bf16_t*)alloc((size_t)Nn[0] * H * sizeof(bf16_t));
  xb[1]  = (bf16_t*)alloc((size_t)Nn[1] * H * sizeof(bf16_t));
  acc[0] = (float*)alloc((size_t)Nn[0] * H * sizeof(float));
  acc[1] = (float*)alloc((size_t)Nn[1] * H * sizeof(float));
  int Nmax = Nn[0] > Nn[1] ? Nn[0] : Nn[1];
  float*  hbuf  = (float*)alloc((size_t)Nmax * H * sizeof(float));
  bf16_t* WTb   = (bf16_t*)alloc((size_t)Lnum * 4 * H * H * sizeof(bf16_t));
  bf16_t* linWb = (bf16_t*)alloc((size_t)OUT * H * sizeof(bf16_t));

  // relation table: src type, dst type, self-loops  (0=drug, 1=dis)
  const int  rsrc[4]  = {0, 1, 0, 1};
  const int  rdst[4]  = {0, 1, 1, 0};
  const bool rself[4] = {true, true, false, false};
  float* dinvS[4]; float* dinvD[4];
  for (int r = 0; r < 4; ++r) {
    dinvS[r] = (float*)alloc((size_t)Nn[rsrc[r]] * sizeof(float));
    dinvD[r] = (float*)alloc((size_t)Nn[rdst[r]] * sizeof(float));
  }

  auto g1 = [](long long n) { return dim3((unsigned)((n + 255) / 256)); };

  // pack weights to bf16 (B pre-transposed for column-major WMMA operand loads)
  {
    int total = Lnum * 4 * H * H;
    k_pack_wt<<<g1(total), 256, 0, stream>>>(Ws, WTb, total);
    k_cvt_bf16<<<g1(OUT * H), 256, 0, stream>>>(lin_w, linWb, OUT * H);
  }

  // symmetric-norm degree factors (once; shared by both layers)
  for (int r = 0; r < 4; ++r) {
    int ns = Nn[rsrc[r]], nd = Nn[rdst[r]];
    k_zero_f32<<<g1(ns), 256, 0, stream>>>(dinvS[r], ns);
    k_zero_f32<<<g1(nd), 256, 0, stream>>>(dinvD[r], nd);
    k_count<<<g1(E), 256, 0, stream>>>(ei[r],     dinvS[r], E);
    k_count<<<g1(E), 256, 0, stream>>>(ei[r] + E, dinvD[r], E);
    float sa = rself[r] ? 1.0f : 0.0f;
    k_dinv<<<g1(ns), 256, 0, stream>>>(dinvS[r], ns, sa);
    k_dinv<<<g1(nd), 256, 0, stream>>>(dinvD[r], nd, sa);
  }

  const float* xcur[2] = { x0[0], x0[1] };
  for (int l = 0; l < Lnum; ++l) {
    for (int t = 0; t < 2; ++t) {
      k_cvt_bf16<<<g1((long long)Nn[t] * H), 256, 0, stream>>>(xcur[t], xb[t], Nn[t] * H);
      k_zero_f32<<<g1((long long)Nn[t] * H), 256, 0, stream>>>(acc[t], Nn[t] * H);
    }
    for (int r = 0; r < 4; ++r) {
      int s = rsrc[r], d = rdst[r];
      const bf16_t* BT = WTb + (size_t)(l * 4 + r) * H * H;
      int tilesM = Nn[s] >> 4, tilesN = H >> 4;
      k_gemm_bf16<<<g1((long long)tilesM * tilesN * 32), 256, 0, stream>>>(
          xb[s], BT, hbuf, nullptr, tilesM, tilesN, H);
      k_scatter<<<g1((long long)E * 32), 256, 0, stream>>>(
          ei[r], ei[r] + E, hbuf, acc[d], dinvS[r], dinvD[r], E);
      if (rself[r])
        k_selfloop<<<g1((long long)Nn[s] * H), 256, 0, stream>>>(
            hbuf, acc[d], dinvS[r], dinvD[r], Nn[s] * H);
    }
    const float* bl = bs + (size_t)l * 4 * H;
    k_relu_bias<<<g1((long long)Nn[0] * H), 256, 0, stream>>>(acc[0], bl + 0 * H, bl + 3 * H, Nn[0] * H);
    k_relu_bias<<<g1((long long)Nn[1] * H), 256, 0, stream>>>(acc[1], bl + 1 * H, bl + 2 * H, Nn[1] * H);
    xcur[0] = acc[0]; xcur[1] = acc[1];
  }

  // final linear to OUT channels, bias fused into WMMA epilogue
  for (int t = 0; t < 2; ++t)
    k_cvt_bf16<<<g1((long long)Nn[t] * H), 256, 0, stream>>>(xcur[t], xb[t], Nn[t] * H);
  float* outp = out;
  for (int t = 0; t < 2; ++t) {
    int tilesM = Nn[t] >> 4, tilesN = OUT >> 4;
    k_gemm_bf16<<<g1((long long)tilesM * tilesN * 32), 256, 0, stream>>>(
        xb[t], linWb, outp, lin_b, tilesM, tilesN, OUT);
    outp += (size_t)Nn[t] * OUT;
  }
}
